// Attention_27943057228498
// MI455X (gfx1250) — matrix operand order repaired and checked by reading, compiled, not run on hardware
//
#include <hip/hip_runtime.h>
#include <hip/hip_bf16.h>

// ---------------- problem constants ----------------
constexpr int CB  = 32;    // batch
constexpr int CN  = 577;   // tokens
constexpr int CNP = 608;   // tokens padded to multiple of 32
constexpr int CC  = 768;   // channels
constexpr int CH  = 12;    // heads
constexpr int CHD = 64;    // head dim
constexpr int CBH = CB * CH;          // 384
constexpr int CCO = 3 * CC;           // 2304 (qkv out)
constexpr int CNK = 462;              // kept tokens (577 - 115)
constexpr int CKT = (CNK + 15) / 16;  // 29 kept-row tiles
constexpr int CRT = (CN + 15) / 16;   // 37 16-col tiles of keys

typedef __attribute__((ext_vector_type(16))) __bf16 v16bf;
typedef __attribute__((ext_vector_type(8)))  float  v8f;

// ---------------- helpers ----------------
__device__ __forceinline__ unsigned short f2bf(float f) {
    unsigned u = __float_as_uint(f);
    unsigned r = (u + 0x7FFFu + ((u >> 16) & 1u)) >> 16;
    return (unsigned short)r;
}

// A-fragment (16-bit, 16x32): lane holds row = lane%16; VGPR j holds K pair at
// k = (j>=4?16:0) + 8*half + 2*(j&3)   (ISA 7.12.2, 16-bit A 16x32)
__device__ __forceinline__ v16bf load_afrag_g(const unsigned short* rowptr, int half) {
    union { v16bf v; unsigned u[8]; } a;
#pragma unroll
    for (int j = 0; j < 8; ++j) {
        int k = ((j >= 4) ? 16 : 0) + half * 8 + (j & 3) * 2;
        a.u[j] = *(const unsigned*)(rowptr + k);
    }
    return a.v;
}

// Same layout, built from u16 loads (safe for same-wave LDS write->read)
__device__ __forceinline__ v16bf load_afrag_lds(const unsigned short* rowptr, int half) {
    union { v16bf v; unsigned u[8]; } a;
#pragma unroll
    for (int j = 0; j < 8; ++j) {
        int k = ((j >= 4) ? 16 : 0) + half * 8 + (j & 3) * 2;
        a.u[j] = (unsigned)rowptr[k] | ((unsigned)rowptr[k + 1] << 16);
    }
    return a.v;
}

// B-fragment (16-bit, 32x16): lane holds col = lane%16; VGPR j holds K pair at
// k = (j>=4?16:0) + 8*half + 2*(j&3) ; colptr must be contiguous along K.
__device__ __forceinline__ v16bf load_bfrag(const unsigned short* colptr, int half) {
    union { v16bf v; unsigned u[8]; } b;
#pragma unroll
    for (int j = 0; j < 8; ++j) {
        int k = ((j >= 4) ? 16 : 0) + half * 8 + (j & 3) * 2;
        b.u[j] = *(const unsigned*)(colptr + k);
    }
    return b.v;
}

#define WMMA_BF16(A, Bm, Cm) \
    __builtin_amdgcn_wmma_f32_16x16x32_bf16(false, (A), false, (Bm), (short)0, (Cm), false, false)

// ---------------- kernels ----------------
__global__ void cvt_f32_bf16_kernel(const float* __restrict__ src,
                                    unsigned short* __restrict__ dst, long n) {
    long i = blockIdx.x * (long)blockDim.x + threadIdx.x;
    long stride = (long)gridDim.x * blockDim.x;
    for (; i < n; i += stride) dst[i] = f2bf(src[i]);
}

// qkv = x @ qkv_w^T, 32x32 register-blocked (4 WMMA per 4 fragment loads);
// scatter into Q(bh,n,d)*scale, K(bh,n,d), V^T(bh,d,n) bf16, zero padded rows.
__global__ void qkv_gemm_kernel(const unsigned short* __restrict__ xbf,
                                const unsigned short* __restrict__ wbf,
                                unsigned short* __restrict__ qbf,
                                unsigned short* __restrict__ kbf,
                                unsigned short* __restrict__ vt) {
    const int lane = threadIdx.x & 31, half = lane >> 4, l16 = lane & 15;
    int wid = blockIdx.x * (blockDim.x >> 5) + (threadIdx.x >> 5);
    const int Ntiles = CCO / 32;                    // 72
    const int total  = (CB * CNP / 32) * Ntiles;    // 43776
    if (wid >= total) return;
    int mt = wid / Ntiles, nt = wid % Ntiles;
    int rowbase = mt * 32, colbase = nt * 32;

    int m0 = rowbase + l16, m1 = rowbase + 16 + l16;
    int b0i = m0 / CNP, n0 = m0 % CNP;
    int b1i = m1 / CNP, n1 = m1 % CNP;
    const unsigned short* aptr0 = xbf + ((size_t)b0i * CN + (n0 < CN ? n0 : CN - 1)) * CC;
    const unsigned short* aptr1 = xbf + ((size_t)b1i * CN + (n1 < CN ? n1 : CN - 1)) * CC;
    const unsigned short* bptr0 = wbf + (size_t)(colbase + l16) * CC;
    const unsigned short* bptr1 = wbf + (size_t)(colbase + 16 + l16) * CC;

    v8f cc[2][2] = {{{}, {}}, {{}, {}}};
    for (int kk = 0; kk < CC; kk += 32) {
        if (kk + 64 <= CC) {                 // prefetch next K blocks
            __builtin_prefetch(aptr0 + kk + 32);
            __builtin_prefetch(bptr0 + kk + 32);
        }
        v16bf a0 = load_afrag_g(aptr0 + kk, half);
        v16bf a1 = load_afrag_g(aptr1 + kk, half);
        v16bf b0 = load_bfrag(bptr0 + kk, half);
        v16bf b1 = load_bfrag(bptr1 + kk, half);
        cc[0][0] = WMMA_BF16(a0, b0, cc[0][0]);
        cc[0][1] = WMMA_BF16(a0, b1, cc[0][1]);
        cc[1][0] = WMMA_BF16(a1, b0, cc[1][0]);
        cc[1][1] = WMMA_BF16(a1, b1, cc[1][1]);
    }

#pragma unroll
    for (int J = 0; J < 2; ++J) {
        int col = colbase + J * 16 + l16;
        int three = col / CC, rem = col % CC, h = rem / CHD, d = rem % CHD;
#pragma unroll
        for (int I = 0; I < 2; ++I) {
#pragma unroll
            for (int i = 0; i < 8; ++i) {
                int mi = rowbase + I * 16 + i + 8 * half;
                int bi = mi / CNP, ni = mi % CNP;
                float v = (ni < CN) ? cc[I][J][i] : 0.0f;
                size_t bh = (size_t)bi * CH + h;
                if (three == 0)      qbf[(bh * CNP + ni) * CHD + d] = f2bf(v * 0.125f);
                else if (three == 1) kbf[(bh * CNP + ni) * CHD + d] = f2bf(v);
                else                 vt [(bh * CHD + d) * CNP + ni] = f2bf(v);
            }
        }
    }
}

// Pass 1: flash-style stats, 32 query rows per wave (B fragments reused 2x).
// Per (bh, 32-row tile): scores q*k^T over all keys, online softmax ->
// rowmax/rowsum; also per-head diagonal score -> prob.
__global__ void attn_pass1_kernel(const unsigned short* __restrict__ qbf,
                                  const unsigned short* __restrict__ kbf,
                                  float* __restrict__ rowmax,
                                  float* __restrict__ rowsum,
                                  float* __restrict__ diag_h) {
    const int lane = threadIdx.x & 31, half = lane >> 4, l16 = lane & 15;
    int wid = blockIdx.x * (blockDim.x >> 5) + (threadIdx.x >> 5);
    const int RT32 = CNP / 32;                    // 19 row tiles of 32
    const int total = CBH * RT32;                 // 7296
    if (wid >= total) return;
    int bh = wid / RT32, rt = wid % RT32, rowbase = rt * 32;

    const unsigned short* aptr0 = qbf + ((size_t)bh * CNP + rowbase + l16) * CHD;
    const unsigned short* aptr1 = qbf + ((size_t)bh * CNP + rowbase + 16 + l16) * CHD;
    v16bf a[2][2];
    a[0][0] = load_afrag_g(aptr0, half);      a[0][1] = load_afrag_g(aptr0 + 32, half);
    a[1][0] = load_afrag_g(aptr1, half);      a[1][1] = load_afrag_g(aptr1 + 32, half);

    float mx[2][8], sm[2][8], dg[2][8];
#pragma unroll
    for (int g = 0; g < 2; ++g)
#pragma unroll
        for (int i = 0; i < 8; ++i) { mx[g][i] = -1e30f; sm[g][i] = 0.0f; dg[g][i] = -1e30f; }

    for (int ntile = 0; ntile < CRT; ++ntile) {
        int col = ntile * 16 + l16;
        bool cv = col < CN;
        const unsigned short* bptr = kbf + ((size_t)bh * CNP + col) * CHD;
        v16bf b0 = load_bfrag(bptr, half);
        v16bf b1 = load_bfrag(bptr + 32, half);
        v8f c0 = {}, c1 = {};
        c0 = WMMA_BF16(a[0][0], b0, c0);
        c0 = WMMA_BF16(a[0][1], b1, c0);
        c1 = WMMA_BF16(a[1][0], b0, c1);
        c1 = WMMA_BF16(a[1][1], b1, c1);
        if (cv) {
#pragma unroll
            for (int i = 0; i < 8; ++i) {
                {   float x = c0[i];
                    float mn = fmaxf(mx[0][i], x);
                    sm[0][i] = sm[0][i] * __expf(mx[0][i] - mn) + __expf(x - mn);
                    mx[0][i] = mn;
                    if (col == rowbase + i + 8 * half) dg[0][i] = x; }
                {   float x = c1[i];
                    float mn = fmaxf(mx[1][i], x);
                    sm[1][i] = sm[1][i] * __expf(mx[1][i] - mn) + __expf(x - mn);
                    mx[1][i] = mn;
                    if (col == rowbase + 16 + i + 8 * half) dg[1][i] = x; }
            }
        }
    }
    // reduce across the 16 lanes (cols) of each half
    for (int off = 1; off < 16; off <<= 1) {
#pragma unroll
        for (int g = 0; g < 2; ++g)
#pragma unroll
            for (int i = 0; i < 8; ++i) {
                float om = __shfl_xor(mx[g][i], off, 32);
                float os = __shfl_xor(sm[g][i], off, 32);
                float od = __shfl_xor(dg[g][i], off, 32);
                float mn = fmaxf(mx[g][i], om);
                sm[g][i] = sm[g][i] * __expf(mx[g][i] - mn) + os * __expf(om - mn);
                mx[g][i] = mn;
                dg[g][i] = fmaxf(dg[g][i], od);
            }
    }
    if (l16 == 0) {
        int b = bh / CH;
#pragma unroll
        for (int g = 0; g < 2; ++g)
#pragma unroll
            for (int i = 0; i < 8; ++i) {
                int r = rowbase + g * 16 + i + 8 * half;
                if (r < CN) {
                    rowmax[(size_t)bh * CNP + r] = mx[g][i];
                    rowsum[(size_t)bh * CNP + r] = sm[g][i];
                    if (r >= 1) {
                        float p = __expf(dg[g][i] - mx[g][i]) / sm[g][i];
                        diag_h[(size_t)bh * (CN - 1) + (r - 1)] = p;
                    }
                }
            }
    }
}

// mean over heads of the diagonal probs
__global__ void diag_mean_kernel(const float* __restrict__ diag_h,
                                 float* __restrict__ diag_m) {
    int i = blockIdx.x * blockDim.x + threadIdx.x;    // over B*(N-1)
    if (i >= CB * (CN - 1)) return;
    int b = i / (CN - 1), t = i % (CN - 1);
    float s = 0.0f;
    for (int h = 0; h < CH; ++h)
        s += diag_h[((size_t)(b * CH + h)) * (CN - 1) + t];
    diag_m[i] = s * (1.0f / CH);
}

// Stable descending rank (ties -> lower index first, matches jnp.argsort(-v)),
// keep rank < 461, compact ascending with CLS(0) prepended.
// LDS fill uses the CDNA5 async global->LDS path (GV addressing, ASYNCcnt).
__global__ void rank_kernel(const float* __restrict__ diag_m,
                            const int* __restrict__ redp,
                            int* __restrict__ index_kept) {
    __shared__ float vals[CN - 1];
    __shared__ int   flags[CN - 1];
    int b = blockIdx.x, i = threadIdx.x;
#if defined(__gfx1250__)
    {
        const float* gsrc = diag_m + b * (CN - 1) + i;
        unsigned lds_off = (unsigned)(size_t)&vals[i];   // generic LDS ptr low 32b = LDS offset
        asm volatile("global_load_async_to_lds_b32 %0, %1, off"
                     :: "v"(lds_off), "v"((unsigned long long)(size_t)gsrc)
                     : "memory");
        asm volatile("s_wait_asynccnt 0x0" ::: "memory");
    }
#else
    vals[i] = diag_m[b * (CN - 1) + i];
#endif
    __syncthreads();
    float vi = vals[i];
    int r = 0;
    for (int j = 0; j < CN - 1; ++j) {
        float vj = vals[j];
        r += (vj > vi) || (vj == vi && j < i);
    }
    int keepn = (CN - 1) - redp[0];   // 461
    flags[i] = (r < keepn) ? 1 : 0;
    __syncthreads();
    if (flags[i]) {
        int pos = 1;
        for (int j = 0; j < i; ++j) pos += flags[j];
        index_kept[b * CNK + pos] = i + 1;
    }
    if (i == 0) index_kept[b * CNK] = 0;
}

// Pass 2: for kept query rows, recompute scores, P = exp(s-m)/sum, P through
// LDS into A-fragment layout, P @ V via WMMA (P fragment reused 4x over d).
__global__ void attn_pass2_kernel(const unsigned short* __restrict__ qbf,
                                  const unsigned short* __restrict__ kbf,
                                  const unsigned short* __restrict__ vt,
                                  const float* __restrict__ rowmax,
                                  const float* __restrict__ rowsum,
                                  const int* __restrict__ index_kept,
                                  unsigned short* __restrict__ aobf) {
    __shared__ __align__(16) unsigned short pbuf[8][16 * 32];
    const int lane = threadIdx.x & 31, half = lane >> 4, l16 = lane & 15;
    const int w = threadIdx.x >> 5;
    int wid = blockIdx.x * (blockDim.x >> 5) + w;
    const int total = CBH * CKT;                    // 11136
    if (wid >= total) return;
    int bh = wid / CKT, kt = wid % CKT;
    int b = bh / CH, h = bh % CH;

    int kr = kt * 16 + l16;
    int tq = (kr < CNK) ? index_kept[b * CNK + kr] : 0;
    const unsigned short* aptr = qbf + ((size_t)bh * CNP + tq) * CHD;
    v16bf a0 = load_afrag_g(aptr, half);
    v16bf a1 = load_afrag_g(aptr + 32, half);

    float mrow[8], rs[8];
    int krr[8];
#pragma unroll
    for (int i = 0; i < 8; ++i) {
        int kri = kt * 16 + i + 8 * half;
        krr[i] = kri;
        int t = (kri < CNK) ? index_kept[b * CNK + kri] : 0;
        mrow[i] = rowmax[(size_t)bh * CNP + t];
        rs[i] = 1.0f / rowsum[(size_t)bh * CNP + t];
    }

    v8f acc[4] = {{}, {}, {}, {}};
    unsigned short* pb = pbuf[w];

    for (int ntile = 0; ntile < CNP / 32; ++ntile) {   // 19
        int nbase = ntile * 32;
#pragma unroll
        for (int s2 = 0; s2 < 2; ++s2) {
            int col = nbase + s2 * 16 + l16;
            const unsigned short* bptr = kbf + ((size_t)bh * CNP + col) * CHD;
            v16bf b0 = load_bfrag(bptr, half);
            v16bf b1 = load_bfrag(bptr + 32, half);
            v8f c = {};
            c = WMMA_BF16(a0, b0, c);
            c = WMMA_BF16(a1, b1, c);
            bool cvld = col < CN;
#pragma unroll
            for (int i = 0; i < 8; ++i) {
                float p = cvld ? __expf(c[i] - mrow[i]) * rs[i] : 0.0f;
                pb[(i + 8 * half) * 32 + s2 * 16 + l16] = f2bf(p);
            }
        }
        v16bf pf = load_afrag_lds(pb + l16 * 32, half);
#pragma unroll
        for (int dt = 0; dt < 4; ++dt) {
            int d = dt * 16 + l16;
            const unsigned short* vptr = vt + ((size_t)bh * CHD + d) * CNP + nbase;
            v16bf bv = load_bfrag(vptr, half);
            acc[dt] = WMMA_BF16(pf, bv, acc[dt]);
        }
    }
#pragma unroll
    for (int dt = 0; dt < 4; ++dt) {
#pragma unroll
        for (int i = 0; i < 8; ++i) {
            if (krr[i] < CNK)
                aobf[((size_t)b * CNK + krr[i]) * CC + h * CHD + dt * 16 + l16] =
                    f2bf(acc[dt][i]);
        }
    }
}

// out = attn_out @ proj_w^T + proj_b + gather(x, index_kept), 32x32 blocked.
__global__ void proj_gemm_kernel(const unsigned short* __restrict__ aobf,
                                 const unsigned short* __restrict__ wbf,
                                 const float* __restrict__ proj_b,
                                 const float* __restrict__ x,
                                 const int* __restrict__ index_kept,
                                 float* __restrict__ out) {
    const int lane = threadIdx.x & 31, half = lane >> 4, l16 = lane & 15;
    int wid = blockIdx.x * (blockDim.x >> 5) + (threadIdx.x >> 5);
    const int Ntiles = CC / 32;                       // 24
    const int total  = (CB * CNK / 32) * Ntiles;      // 11088
    if (wid >= total) return;
    int mt = wid / Ntiles, nt = wid % Ntiles;
    int rowbase = mt * 32, colbase = nt * 32;

    const unsigned short* aptr0 = aobf + (size_t)(rowbase + l16) * CC;
    const unsigned short* aptr1 = aobf + (size_t)(rowbase + 16 + l16) * CC;
    const unsigned short* bptr0 = wbf + (size_t)(colbase + l16) * CC;
    const unsigned short* bptr1 = wbf + (size_t)(colbase + 16 + l16) * CC;

    v8f cc[2][2] = {{{}, {}}, {{}, {}}};
    for (int kk = 0; kk < CC; kk += 32) {
        if (kk + 64 <= CC) {
            __builtin_prefetch(aptr0 + kk + 32);
            __builtin_prefetch(bptr0 + kk + 32);
        }
        v16bf a0 = load_afrag_g(aptr0 + kk, half);
        v16bf a1 = load_afrag_g(aptr1 + kk, half);
        v16bf b0 = load_bfrag(bptr0 + kk, half);
        v16bf b1 = load_bfrag(bptr1 + kk, half);
        cc[0][0] = WMMA_BF16(a0, b0, cc[0][0]);
        cc[0][1] = WMMA_BF16(a0, b1, cc[0][1]);
        cc[1][0] = WMMA_BF16(a1, b0, cc[1][0]);
        cc[1][1] = WMMA_BF16(a1, b1, cc[1][1]);
    }
#pragma unroll
    for (int J = 0; J < 2; ++J) {
        int col = colbase + J * 16 + l16;
        float pbv = proj_b[col];
#pragma unroll
        for (int I = 0; I < 2; ++I) {
#pragma unroll
            for (int i = 0; i < 8; ++i) {
                int m = rowbase + I * 16 + i + 8 * half;
                int b = m / CNK, kr = m % CNK;
                int t = index_kept[b * CNK + kr];
                float xv = x[((size_t)b * CN + t) * CC + col];
                out[(size_t)m * CC + col] = cc[I][J][i] + pbv + xv;
            }
        }
    }
}

// ---------------- host launcher ----------------
extern "C" void kernel_launch(void* const* d_in, const int* in_sizes, int n_in,
                              void* d_out, int out_size, void* d_ws, size_t ws_size,
                              hipStream_t stream) {
    (void)in_sizes; (void)n_in; (void)out_size; (void)ws_size;
    const float* x      = (const float*)d_in[0];
    const float* qkv_w  = (const float*)d_in[1];
    const float* proj_w = (const float*)d_in[2];
    const float* proj_b = (const float*)d_in[3];
    const int*   redp   = (const int*)d_in[4];
    float* out = (float*)d_out;

    char* wsp = (char*)d_ws;
    size_t off = 0;
    auto take = [&](size_t bytes) -> void* {
        void* p = wsp + off;
        off += (bytes + 255) & ~(size_t)255;
        return p;
    };
    unsigned short* xbf    = (unsigned short*)take((size_t)CB * CN * CC * 2);
    unsigned short* wqkvbf = (unsigned short*)take((size_t)CCO * CC * 2);
    unsigned short* wprjbf = (unsigned short*)take((size_t)CC * CC * 2);
    unsigned short* qbf    = (unsigned short*)take((size_t)CBH * CNP * CHD * 2);
    unsigned short* kbf    = (unsigned short*)take((size_t)CBH * CNP * CHD * 2);
    unsigned short* vt     = (unsigned short*)take((size_t)CBH * CHD * CNP * 2);
    float* rowmax  = (float*)take((size_t)CBH * CNP * 4);
    float* rowsum  = (float*)take((size_t)CBH * CNP * 4);
    float* diag_h  = (float*)take((size_t)CBH * (CN - 1) * 4);
    float* diag_m  = (float*)take((size_t)CB * (CN - 1) * 4);
    int*   idxk    = (int*)take((size_t)CB * CNK * 4);
    unsigned short* aobf = (unsigned short*)take((size_t)CB * CNK * CC * 2);

    // 1) fp32 -> bf16 conversions
    cvt_f32_bf16_kernel<<<2048, 256, 0, stream>>>(x, xbf, (long)CB * CN * CC);
    cvt_f32_bf16_kernel<<<1024, 256, 0, stream>>>(qkv_w, wqkvbf, (long)CCO * CC);
    cvt_f32_bf16_kernel<<<512, 256, 0, stream>>>(proj_w, wprjbf, (long)CC * CC);

    // 2) QKV GEMM (32x32 blocked): 43776 wave-tiles, 8 waves/block
    qkv_gemm_kernel<<<5472, 256, 0, stream>>>(xbf, wqkvbf, qbf, kbf, vt);

    // 3) flash pass 1 (32 rows/wave): 7296 wave-tiles
    attn_pass1_kernel<<<912, 256, 0, stream>>>(qbf, kbf, rowmax, rowsum, diag_h);

    // 4) mean over heads, then rank + compact kept indices
    diag_mean_kernel<<<(CB * (CN - 1) + 255) / 256, 256, 0, stream>>>(diag_h, diag_m);
    rank_kernel<<<CB, CN - 1, 0, stream>>>(diag_m, redp, idxk);

    // 5) flash pass 2: P @ V for kept rows: 11136 wave-tiles
    attn_pass2_kernel<<<1392, 256, 0, stream>>>(qbf, kbf, vt, rowmax, rowsum, idxk, aobf);

    // 6) projection GEMM (32x32 blocked) + bias + gathered residual
    proj_gemm_kernel<<<1386, 256, 0, stream>>>(aobf, wprjbf, proj_b, x, idxk, out);
}
